// CSNA_4337916969344
// MI455X (gfx1250) — matrix operand order repaired and checked by reading, not compiled, not run on hardware
//
#include <hip/hip_runtime.h>
#include <math.h>

typedef __attribute__((ext_vector_type(16))) __bf16         v16bf;
typedef __attribute__((ext_vector_type(8)))  float          v8f;
typedef __attribute__((ext_vector_type(16))) unsigned short v16u;
typedef __attribute__((ext_vector_type(8)))  unsigned short v8u;

#define H_DIM 128
#define BSTRIDE 136   // LDS row stride in ushorts: 272B rows, 16B aligned, bank-skewed

// ---------------- helpers ----------------
__device__ __forceinline__ float wave_sum(float v) {
    #pragma unroll
    for (int m = 16; m; m >>= 1) v += __shfl_xor(v, m, 32);
    return v;
}

__device__ __forceinline__ v16u combine8(v8u lo, v8u hi) {
    v16u r;
    #pragma unroll
    for (int i = 0; i < 8; ++i) { r[i] = lo[i]; r[8 + i] = hi[i]; }
    return r;
}

// Async global->LDS stage of a [Nc,128] bf16 weight into skewed LDS rows.
// Uses gfx1250 GLOBAL_LOAD_ASYNC_TO_LDS_B128 (ASYNCcnt) per 16B chunk.
__device__ __forceinline__ void stage_weight_async(const unsigned short* __restrict__ Wp,
                                                   unsigned short* sB, int tid,
                                                   int nrows, int Nc) {
    for (int g = tid; g < nrows * 16; g += 256) {
        int row = g >> 4, cb = (g & 15) * 8;
        unsigned short* dst = sB + row * BSTRIDE + cb;
        if (row < Nc) {
            // addrspace(3) cast -> 32-bit LDS byte offset
            unsigned lds = (unsigned)(size_t)(__attribute__((address_space(3))) unsigned short*)dst;
            unsigned long long ga = (unsigned long long)(size_t)(Wp + (size_t)row * H_DIM + cb);
            asm volatile("global_load_async_to_lds_b128 %0, %1, off"
                         :: "v"(lds), "v"(ga) : "memory");
        } else {
            v8u z = {};
            *(v8u*)dst = z;  // zero-pad tail rows (classifier Nc=40)
        }
    }
    asm volatile("s_wait_asynccnt 0x0" ::: "memory");
}

// ---------------- fp32 -> bf16 (RNE) ----------------
__global__ void k_f32_to_bf16(const float* __restrict__ src, unsigned short* __restrict__ dst, size_t n) {
    size_t i = (size_t)blockIdx.x * blockDim.x + threadIdx.x;
    if (i >= n) return;
    unsigned u = __float_as_uint(src[i]);
    u += 0x7FFFu + ((u >> 16) & 1u);
    dst[i] = (unsigned short)(u >> 16);
}

// ---------------- WMMA GEMM, weight staged in LDS ----------------
// C[M,Nc] = A[M,128] @ W[Nc,128]^T (+bias). One wave per 16-row strip of C;
// A fragments held in VGPRs across all column tiles; B fragments read from LDS.
__global__ void k_gemm_lds(const unsigned short* __restrict__ A,
                           const unsigned short* __restrict__ W,
                           const float* __restrict__ bias,
                           float* __restrict__ Cmat,
                           int M, int Nc, int ldc) {
    __shared__ unsigned short sB[H_DIM * BSTRIDE];
    const int ntiles = (Nc + 15) >> 4;
    const int tid = threadIdx.y * 32 + threadIdx.x;

    stage_weight_async(W, sB, tid, ntiles * 16, Nc);
    __syncthreads();

    const int tm_count = (M + 15) >> 4;
    const int tm = blockIdx.x * blockDim.y + threadIdx.y;
    const bool active = tm < tm_count;
    const int tmc = active ? tm : tm_count - 1;
    const int lane = threadIdx.x, half = lane >> 4, l16 = lane & 15;

    // A fragments for the whole K=128 strip, kept in VGPRs
    const unsigned short* arp = A + ((size_t)tmc * 16 + l16) * H_DIM;
    v16u afrag[4];
    #pragma unroll
    for (int kk = 0; kk < 4; ++kk) {
        v8u lo = *(const v8u*)(arp + kk * 32 + half * 8);
        v8u hi = *(const v8u*)(arp + kk * 32 + 16 + half * 8);
        afrag[kk] = combine8(lo, hi);
    }

    for (int tn = 0; tn < ntiles; ++tn) {
        const unsigned short* bp = sB + (tn * 16 + l16) * BSTRIDE;
        v8f acc = {};
        #pragma unroll
        for (int kk = 0; kk < 4; ++kk) {
            v8u blo = *(const v8u*)(bp + kk * 32 + half * 8);
            v8u bhi = *(const v8u*)(bp + kk * 32 + 16 + half * 8);
            acc = __builtin_amdgcn_wmma_f32_16x16x32_bf16(
                false, __builtin_bit_cast(v16bf, afrag[kk]),
                false, __builtin_bit_cast(v16bf, combine8(blo, bhi)),
                (short)0, acc, false, false);
        }
        if (active) {
            int col = tn * 16 + l16;
            if (col < Nc) {
                float bb = bias ? bias[col] : 0.f;
                #pragma unroll
                for (int r = 0; r < 8; ++r)
                    Cmat[(size_t)(tm * 16 + half * 8 + r) * ldc + col] = acc[r] + bb;
            }
        }
    }
}

// ---------------- fused 4-matrix WMMA GEMM (shared A operand) ----------------
// o_w = A @ W_w^T for w=0..3 (bias only on w==3: Wself). A fragments loaded once.
__global__ void k_gemm4_lds(const unsigned short* __restrict__ A,
                            const unsigned short* __restrict__ W0,
                            const unsigned short* __restrict__ W1,
                            const unsigned short* __restrict__ W2,
                            const unsigned short* __restrict__ W3,
                            const float* __restrict__ bias3,
                            float* __restrict__ o0, float* __restrict__ o1,
                            float* __restrict__ o2, float* __restrict__ o3,
                            int M) {
    __shared__ unsigned short sB[H_DIM * BSTRIDE];
    const unsigned short* Ws[4] = {W0, W1, W2, W3};
    float* Os[4] = {o0, o1, o2, o3};
    const int tid = threadIdx.y * 32 + threadIdx.x;

    const int tm_count = (M + 15) >> 4;
    const int tm = blockIdx.x * blockDim.y + threadIdx.y;
    const bool active = tm < tm_count;
    const int tmc = active ? tm : tm_count - 1;
    const int lane = threadIdx.x, half = lane >> 4, l16 = lane & 15;

    const unsigned short* arp = A + ((size_t)tmc * 16 + l16) * H_DIM;
    v16u afrag[4];
    #pragma unroll
    for (int kk = 0; kk < 4; ++kk) {
        v8u lo = *(const v8u*)(arp + kk * 32 + half * 8);
        v8u hi = *(const v8u*)(arp + kk * 32 + 16 + half * 8);
        afrag[kk] = combine8(lo, hi);
    }

    for (int w = 0; w < 4; ++w) {
        __syncthreads();  // previous iteration's readers done
        stage_weight_async(Ws[w], sB, tid, H_DIM, H_DIM);
        __syncthreads();

        float* Op = Os[w];
        const float* bias = (w == 3) ? bias3 : nullptr;
        #pragma unroll
        for (int tn = 0; tn < 8; ++tn) {
            const unsigned short* bp = sB + (tn * 16 + l16) * BSTRIDE;
            v8f acc = {};
            #pragma unroll
            for (int kk = 0; kk < 4; ++kk) {
                v8u blo = *(const v8u*)(bp + kk * 32 + half * 8);
                v8u bhi = *(const v8u*)(bp + kk * 32 + 16 + half * 8);
                acc = __builtin_amdgcn_wmma_f32_16x16x32_bf16(
                    false, __builtin_bit_cast(v16bf, afrag[kk]),
                    false, __builtin_bit_cast(v16bf, combine8(blo, bhi)),
                    (short)0, acc, false, false);
            }
            if (active) {
                int col = tn * 16 + l16;
                float bb = bias ? bias[col] : 0.f;
                #pragma unroll
                for (int r = 0; r < 8; ++r)
                    Op[(size_t)(tm * 16 + half * 8 + r) * H_DIM + col] = acc[r] + bb;
            }
        }
    }
}

// ---------------- BatchNorm ----------------
__global__ void k_bn_reduce(const float* __restrict__ xin, float* __restrict__ sums, int n) {
    int c = threadIdx.x;  // 0..127, coalesced across a row
    float s = 0.f, s2 = 0.f;
    for (int i = blockIdx.x; i < n; i += gridDim.x) {
        float v = xin[(size_t)i * H_DIM + c];
        s += v; s2 += v * v;
    }
    atomicAdd(&sums[c], s);
    atomicAdd(&sums[H_DIM + c], s2);
}

__global__ void k_bn_apply(const float* __restrict__ xin, const float* __restrict__ sums,
                           const float* __restrict__ gamma, const float* __restrict__ beta,
                           const float* __restrict__ res, float* __restrict__ out,
                           int n, int do_relu) {
    size_t idx = (size_t)blockIdx.x * blockDim.x + threadIdx.x;
    size_t total = (size_t)n * H_DIM;
    if (idx >= total) return;
    int c = (int)(idx & (H_DIM - 1));
    float invn = 1.f / (float)n;
    float m = sums[c] * invn;
    float var = sums[H_DIM + c] * invn - m * m;
    float o = gamma[c] * (xin[idx] - m) * rsqrtf(var + 1e-5f) + beta[c];
    if (do_relu) o = fmaxf(o, 0.f);
    if (res) o += res[idx];
    out[idx] = o;
}

// ---------------- per-node dots with Wh (hoists per-edge h_ij work) ----------------
__global__ void k_node_dots(const float* __restrict__ xg, const float* __restrict__ wh,
                            float* __restrict__ ar, float* __restrict__ ac, int n) {
    int i = blockIdx.x * blockDim.y + threadIdx.y;
    if (i >= n) return;
    int lane = threadIdx.x;
    float4 v  = *(const float4*)(xg + (size_t)i * H_DIM + lane * 4);
    float4 wA = *(const float4*)(wh + lane * 4);
    float4 wB = *(const float4*)(wh + H_DIM + lane * 4);
    float pa = v.x * wA.x + v.y * wA.y + v.z * wA.z + v.w * wA.w;
    float pb = v.x * wB.x + v.y * wB.y + v.z * wB.z + v.w * wB.w;
    pa = wave_sum(pa);
    pb = wave_sum(pb);
    if (lane == 0) { ar[i] = pa; ac[i] = pb; }
}

// ---------------- edge pass 1: s_ij + softmax denominators ----------------
__global__ void k_edge_pass1(const float* __restrict__ xg,
                             const float* __restrict__ ar, const float* __restrict__ ac,
                             const float* __restrict__ bh, const int* __restrict__ ei,
                             float* __restrict__ s_e,
                             float* __restrict__ z_con, float* __restrict__ z_dis,
                             int E, int n) {
    int e = blockIdx.x * blockDim.y + threadIdx.y;
    if (e >= E + n) return;
    int r, c;
    if (e < E) { r = ei[e]; c = ei[E + e]; } else { r = c = e - E; }
    int lane = threadIdx.x;
    float4 xr4 = *(const float4*)(xg + (size_t)r * H_DIM + lane * 4);
    float4 xc4 = *(const float4*)(xg + (size_t)c * H_DIM + lane * 4);
    float d0 = xr4.x - xc4.x, d1 = xr4.y - xc4.y, d2 = xr4.z - xc4.z, d3 = xr4.w - xc4.w;
    float ss = wave_sum(d0 * d0 + d1 * d1 + d2 * d2 + d3 * d3);
    if (lane == 0) {
        float g  = sqrtf(ss + 1e-12f);
        float hx = ar[r] + ac[c] + bh[0];
        float h  = (hx > 20.f) ? hx : log1pf(expf(hx));
        float s  = 1.f / (1.f + expf(g + h));  // sigmoid(-(g+h)/tau), tau=1
        s_e[e] = s;
        atomicAdd(&z_con[r], expf(s));    // softmax(s): shift-invariant, s in (0,1)
        atomicAdd(&z_dis[r], expf(-s));   // softmax(1-s) == softmax(-s)
    }
}

// ---------------- edge pass 2: weighted scatter-add ----------------
__global__ void k_edge_pass2(const float* __restrict__ xcon, const float* __restrict__ xdis,
                             const float* __restrict__ s_e,
                             const float* __restrict__ z_con, const float* __restrict__ z_dis,
                             const int* __restrict__ ei,
                             float* __restrict__ ocon, float* __restrict__ odis,
                             int E, int n) {
    int e = blockIdx.x * blockDim.y + threadIdx.y;
    if (e >= E + n) return;
    int r, c;
    if (e < E) { r = ei[e]; c = ei[E + e]; } else { r = c = e - E; }
    int lane = threadIdx.x;
    float s  = s_e[e];
    float wc = expf(s)  / z_con[r];
    float wd = expf(-s) / z_dis[r];
    float4 vc = *(const float4*)(xcon + (size_t)r * H_DIM + lane * 4);
    float4 vd = *(const float4*)(xdis + (size_t)r * H_DIM + lane * 4);
    float* oc = ocon + (size_t)c * H_DIM + lane * 4;
    float* od = odis + (size_t)c * H_DIM + lane * 4;
    atomicAdd(oc + 0, wc * vc.x); atomicAdd(oc + 1, wc * vc.y);
    atomicAdd(oc + 2, wc * vc.z); atomicAdd(oc + 3, wc * vc.w);
    atomicAdd(od + 0, wd * vd.x); atomicAdd(od + 1, wd * vd.y);
    atomicAdd(od + 2, wd * vd.z); atomicAdd(od + 3, wd * vd.w);
}

// ---------------- gate: 3-way softmax mixture (+optional residual) ----------------
__global__ void k_gate(const float* __restrict__ con, const float* __restrict__ dis,
                       const float* __restrict__ slf,
                       const float* __restrict__ Wgate, const float* __restrict__ bgate,
                       const float* __restrict__ res, float* __restrict__ out, int n) {
    int i = blockIdx.x * blockDim.y + threadIdx.y;
    if (i >= n) return;
    int lane = threadIdx.x;
    size_t base = (size_t)i * H_DIM;
    float p0 = 0.f, p1 = 0.f, p2 = 0.f;
    #pragma unroll
    for (int j = 0; j < 4; ++j) {
        int ch = lane * 4 + j;
        float v0 = con[base + ch], v1 = dis[base + ch], v2 = slf[base + ch];
        p0 += v0 * Wgate[ch]       + v1 * Wgate[128 + ch]       + v2 * Wgate[256 + ch];
        p1 += v0 * Wgate[384 + ch] + v1 * Wgate[384 + 128 + ch] + v2 * Wgate[384 + 256 + ch];
        p2 += v0 * Wgate[768 + ch] + v1 * Wgate[768 + 128 + ch] + v2 * Wgate[768 + 256 + ch];
    }
    p0 = wave_sum(p0) + bgate[0];
    p1 = wave_sum(p1) + bgate[1];
    p2 = wave_sum(p2) + bgate[2];
    float m = fmaxf(p0, fmaxf(p1, p2));
    float e0 = expf(p0 - m), e1 = expf(p1 - m), e2 = expf(p2 - m);
    float inv = 1.f / (e0 + e1 + e2);
    float g0 = e0 * inv, g1 = e1 * inv, g2 = e2 * inv;
    #pragma unroll
    for (int j = 0; j < 4; ++j) {
        int ch = lane * 4 + j;
        float o = g0 * con[base + ch] + g1 * dis[base + ch] + g2 * slf[base + ch];
        if (res) o += res[base + ch];
        out[base + ch] = o;
    }
}

// ---------------- host orchestration ----------------
extern "C" void kernel_launch(void* const* d_in, const int* in_sizes, int n_in,
                              void* d_out, int out_size, void* d_ws, size_t ws_size,
                              hipStream_t stream) {
    (void)n_in; (void)out_size; (void)ws_size;
    const float* x     = (const float*)d_in[0];
    const int*   ei    = (const int*)d_in[1];
    const float* mlpW  = (const float*)d_in[2];
    const float* mlpb  = (const float*)d_in[3];
    const float* mlpg  = (const float*)d_in[4];
    const float* mlpbe = (const float*)d_in[5];
    const float* bn0g  = (const float*)d_in[6];
    const float* bn0be = (const float*)d_in[7];
    const float* clsW  = (const float*)d_in[8];
    const float* clsb  = (const float*)d_in[9];
    const float* Wg[2]    = {(const float*)d_in[10], (const float*)d_in[19]};
    const float* Wh[2]    = {(const float*)d_in[11], (const float*)d_in[20]};
    const float* bh[2]    = {(const float*)d_in[12], (const float*)d_in[21]};
    const float* Wcon[2]  = {(const float*)d_in[13], (const float*)d_in[22]};
    const float* Wdis[2]  = {(const float*)d_in[14], (const float*)d_in[23]};
    const float* Wself[2] = {(const float*)d_in[15], (const float*)d_in[24]};
    const float* bself[2] = {(const float*)d_in[16], (const float*)d_in[25]};
    const float* Wgt[2]   = {(const float*)d_in[17], (const float*)d_in[26]};
    const float* bgt[2]   = {(const float*)d_in[18], (const float*)d_in[27]};

    const int N = in_sizes[0] / H_DIM;
    const int E = in_sizes[1] / 2;
    const int C = 40;
    const size_t NB = (size_t)N * H_DIM;

    // workspace layout
    float* f = (float*)d_ws;
    float* h0 = f; f += NB;
    float* x1 = f; f += NB;
    float* x2 = f; f += NB;
    float* t0 = f; f += NB;   // x_g  -> out_con accumulator
    float* t1 = f; f += NB;   // x_con -> gate output (layer 0)
    float* t2 = f; f += NB;   // x_dis
    float* t3 = f; f += NB;   // out_self
    float* t4 = f; f += NB;   // out_dis accumulator
    float* ar  = f; f += N;
    float* ac  = f; f += N;
    float* zc  = f; f += N;
    float* zd  = f; f += N;
    float* se  = f; f += (size_t)E + N;
    float* bns = f; f += 256;
    unsigned short* ub    = (unsigned short*)f;
    unsigned short* xbf   = ub; ub += NB;
    unsigned short* wbMlp = ub; ub += H_DIM * H_DIM;
    unsigned short* wbCls = ub; ub += C * H_DIM;
    unsigned short* wbG[2], *wbC[2], *wbD[2], *wbS[2];
    for (int l = 0; l < 2; ++l) {
        wbG[l] = ub; ub += H_DIM * H_DIM;
        wbC[l] = ub; ub += H_DIM * H_DIM;
        wbD[l] = ub; ub += H_DIM * H_DIM;
        wbS[l] = ub; ub += H_DIM * H_DIM;
    }

    const dim3 wblk(32, 8);
    const int HH = H_DIM * H_DIM;
    const unsigned cvt_nb   = (unsigned)((NB + 255) / 256);
    const unsigned cvt_hh   = (unsigned)((HH + 255) / 256);
    const unsigned nodeGrid = (unsigned)((N + 7) / 8);
    const unsigned edgeGrid = (unsigned)(((size_t)E + N + 7) / 8);
    const unsigned gemmGrid = (unsigned)(((N + 15) / 16 + 7) / 8);  // 16-row strips, 8 waves/block
    const unsigned ewGrid   = (unsigned)((NB + 255) / 256);

    // --- weight conversions (fp32 -> bf16) ---
    k_f32_to_bf16<<<cvt_hh, 256, 0, stream>>>(mlpW, wbMlp, (size_t)HH);
    k_f32_to_bf16<<<(unsigned)((C * H_DIM + 255) / 256), 256, 0, stream>>>(clsW, wbCls, (size_t)C * H_DIM);
    for (int l = 0; l < 2; ++l) {
        k_f32_to_bf16<<<cvt_hh, 256, 0, stream>>>(Wg[l],    wbG[l], (size_t)HH);
        k_f32_to_bf16<<<cvt_hh, 256, 0, stream>>>(Wcon[l],  wbC[l], (size_t)HH);
        k_f32_to_bf16<<<cvt_hh, 256, 0, stream>>>(Wdis[l],  wbD[l], (size_t)HH);
        k_f32_to_bf16<<<cvt_hh, 256, 0, stream>>>(Wself[l], wbS[l], (size_t)HH);
    }

    // --- input MLP + BN + ReLU -> h0 ---
    k_f32_to_bf16<<<cvt_nb, 256, 0, stream>>>(x, xbf, NB);
    k_gemm_lds<<<gemmGrid, wblk, 0, stream>>>(xbf, wbMlp, mlpb, t1, N, H_DIM, H_DIM);
    hipMemsetAsync(bns, 0, 256 * sizeof(float), stream);
    k_bn_reduce<<<512, H_DIM, 0, stream>>>(t1, bns, N);
    k_bn_apply<<<ewGrid, 256, 0, stream>>>(t1, bns, mlpg, mlpbe, nullptr, h0, N, 1);
    k_f32_to_bf16<<<cvt_nb, 256, 0, stream>>>(h0, xbf, NB);

    // --- two CSNA conv layers ---
    for (int l = 0; l < 2; ++l) {
        // fused: x_g, x_con, x_dis, out_self (shared A operand, weights cycled through LDS)
        k_gemm4_lds<<<gemmGrid, wblk, 0, stream>>>(xbf, wbG[l], wbC[l], wbD[l], wbS[l],
                                                   bself[l], t0, t1, t2, t3, N);
        k_node_dots<<<nodeGrid, wblk, 0, stream>>>(t0, Wh[l], ar, ac, N);
        hipMemsetAsync(zc, 0, (size_t)N * sizeof(float), stream);
        hipMemsetAsync(zd, 0, (size_t)N * sizeof(float), stream);
        k_edge_pass1<<<edgeGrid, wblk, 0, stream>>>(t0, ar, ac, bh[l], ei, se, zc, zd, E, N);
        // scatter aggregation (x_g in t0 no longer needed -> becomes out_con)
        hipMemsetAsync(t0, 0, NB * sizeof(float), stream);
        hipMemsetAsync(t4, 0, NB * sizeof(float), stream);
        k_edge_pass2<<<edgeGrid, wblk, 0, stream>>>(t1, t2, se, zc, zd, ei, t0, t4, E, N);
        if (l == 0) {
            k_gate<<<nodeGrid, wblk, 0, stream>>>(t0, t4, t3, Wgt[0], bgt[0], nullptr, t1, N);
            hipMemsetAsync(bns, 0, 256 * sizeof(float), stream);
            k_bn_reduce<<<512, H_DIM, 0, stream>>>(t1, bns, N);
            k_bn_apply<<<ewGrid, 256, 0, stream>>>(t1, bns, bn0g, bn0be, h0, x1, N, 1);
            k_f32_to_bf16<<<cvt_nb, 256, 0, stream>>>(x1, xbf, NB);
        } else {
            k_gate<<<nodeGrid, wblk, 0, stream>>>(t0, t4, t3, Wgt[1], bgt[1], x1, x2, N);
            k_f32_to_bf16<<<cvt_nb, 256, 0, stream>>>(x2, xbf, NB);
        }
    }

    // --- classifier ---
    k_gemm_lds<<<gemmGrid, wblk, 0, stream>>>(xbf, wbCls, clsb, (float*)d_out, N, C, C);
}
